// GNN_44633300140131
// MI455X (gfx1250) — matrix operand order repaired and checked by reading, not compiled, not run on hardware
//
#include <hip/hip_runtime.h>
#include <hip/hip_bf16.h>
#include <math.h>

// ---------------------------------------------------------------------------
// HGT (2 layers) for MI455X / gfx1250, wave32 + WMMA + async LDS-DMA.
// Dense work (all projections incl. fused relation transforms) runs on
// v_wmma_f32_16x16x32_bf16 with B staged into LDS via
// global_load_async_to_lds_b128; edge phase is bandwidth-bound gather/scatter
// with native f32 global atomics for the segment softmax / segment sum.
// ---------------------------------------------------------------------------

#define DIM   128
#define HEADS 8
#define HD    16
#define CDIV(a, b) (((a) + (b) - 1) / (b))

typedef __attribute__((ext_vector_type(16))) __bf16 v16bf;
typedef __attribute__((ext_vector_type(8)))  float  v8f;

union BF16x16 { uint4 q[2]; v16bf v; };

__device__ inline __bf16 f2bf(float f) {           // round-to-nearest-even f32->bf16
  unsigned u = __float_as_uint(f);
  unsigned r = (u + 0x7FFFu + ((u >> 16) & 1u)) >> 16;
  unsigned short hs = (unsigned short)r;
  __bf16 o; __builtin_memcpy(&o, &hs, 2); return o;
}

__device__ inline void atomicMaxF(float* addr, float v) {
  // monotone f32 max built from integer atomics (works with -inf init)
  if (v >= 0.f) atomicMax((int*)addr, __float_as_int(v));
  else          atomicMin((unsigned int*)addr, __float_as_uint(v));
}

// ---------------------------------------------------------------------------
// WMMA GEMM: C[nrows x 128] = A(bf16)[nrows x 128] @ B[128 x 128] + bias
// B is passed PRE-TRANSPOSED: Bt[col][k] row-major (bf16).
// optional epilogue: C = s*C + (1-s)*xold with s = sigmoid(*skip)
// Block: 128 threads (4 waves); each wave -> 16 rows x 128 cols (8 WMMA tiles).
// Bt is staged into LDS with async LDS-DMA (ASYNCcnt), no VGPR round-trip.
// ---------------------------------------------------------------------------
__global__ __launch_bounds__(128)
void hgt_gemm128(const __bf16* __restrict__ Bt_, const __bf16* __restrict__ A,
                 const float* __restrict__ bias, float* __restrict__ C, int nrows,
                 const float* __restrict__ xold, const float* __restrict__ skip)
{
  __shared__ __bf16 sBt[DIM][DIM + 8];   // stride 136 halves = bank-conflict-free reads

  // ---- stage Bt into LDS via global_load_async_to_lds_b128 (16B per lane) ----
  {
    const unsigned ldsBase = (unsigned)(size_t)(&sBt[0][0]);  // low 32 bits = LDS offset
    const unsigned tid = threadIdx.x;
    #pragma unroll
    for (int it = 0; it < 16; ++it) {
      unsigned f = tid + 128u * it;          // 16-byte chunk id: 2048 chunks = 32KB
      unsigned c = f >> 4, i = f & 15u;      // row (col-of-B), chunk within row
      unsigned lds = ldsBase + c * 272u + i * 16u;             // stride 136 halves
      const char* g = (const char*)Bt_ + (size_t)c * 256u + i * 16u;
      asm volatile("global_load_async_to_lds_b128 %0, %1, off"
                   :: "v"(lds), "v"(g) : "memory");
    }
    asm volatile("s_wait_asynccnt 0" ::: "memory");
  }
  __syncthreads();

  const int lane = threadIdx.x & 31;
  const int wave = threadIdx.x >> 5;
  const int r0   = blockIdx.x * 64 + wave * 16;
  if (r0 >= nrows) return;

  const int hl  = (lane < 16) ? 0 : 1;   // half-wave select
  const int rIn = lane & 15;
  int row = r0 + rIn; if (row >= nrows) row = nrows - 1;   // clamp loads; stores guarded
  const __bf16* arow = A + (size_t)row * DIM;

  v8f acc[8];
  #pragma unroll
  for (int j = 0; j < 8; ++j)
    #pragma unroll
    for (int r = 0; r < 8; ++r) acc[j][r] = 0.f;

  #pragma unroll
  for (int k0 = 0; k0 < DIM; k0 += 32) {
    // A fragment, 16-bit 16x32 layout: halves 0-7 -> K=k0+sel.., halves 8-15 -> K=k0+16+sel..
    BF16x16 af;
    const int sel = hl * 8;
    af.q[0] = *reinterpret_cast<const uint4*>(arow + k0 + sel);
    af.q[1] = *reinterpret_cast<const uint4*>(arow + k0 + 16 + sel);
    #pragma unroll
    for (int j = 0; j < 8; ++j) {
      // B fragment: lane holds col = j*16+rIn, K = k0 + hl*8 + [0..7], k0 + 16 + hl*8 + [0..7]
      BF16x16 bfr;
      const int col = j * 16 + rIn;
      const int kb  = k0 + hl * 8;
      bfr.q[0] = *reinterpret_cast<const uint4*>(&sBt[col][kb]);
      bfr.q[1] = *reinterpret_cast<const uint4*>(&sBt[col][kb + 16]);
      acc[j] = __builtin_amdgcn_wmma_f32_16x16x32_bf16(false, af.v, false, bfr.v,
                                                       (short)0, acc[j], false, false);
    }
  }

  float s = 1.f;
  if (xold) s = 1.f / (1.f + __expf(-skip[0]));
  #pragma unroll
  for (int j = 0; j < 8; ++j) {
    const int col = j * 16 + rIn;
    const float bv = bias[col];
    #pragma unroll
    for (int r = 0; r < 8; ++r) {
      const int gr = r0 + r + hl * 8;       // C/D layout: VGPR r -> M=r (lanes0-15) / M=8+r
      if (gr < nrows) {
        float val = acc[j][r] + bv;
        if (xold) val = s * val + (1.f - s) * xold[(size_t)gr * DIM + col];
        C[(size_t)gr * DIM + col] = val;
      }
    }
  }
}

// ---------------------------------------------------------------------------
// small helpers
// ---------------------------------------------------------------------------
__global__ void fill_f32(float* __restrict__ p, float v, int n) {
  int i = blockIdx.x * blockDim.x + threadIdx.x;
  if (i < n) p[i] = v;
}

// plain convert (for activations, consumed row-major as WMMA A matrix)
__global__ void conv_bf16(__bf16* __restrict__ o, const float* __restrict__ x, int n) {
  int i = blockIdx.x * blockDim.x + threadIdx.x;
  if (i < n) o[i] = f2bf(x[i]);
}

// convert + transpose 128x128 weight: out[c*128+k] = bf16(in[k*128+c])
__global__ void conv_bf16_t(__bf16* __restrict__ o, const float* __restrict__ x) {
  int idx = blockIdx.x * blockDim.x + threadIdx.x;
  if (idx >= DIM * DIM) return;
  int c = idx >> 7, k = idx & 127;
  o[idx] = f2bf(x[k * DIM + c]);
}

// Wc_t = transpose(W @ blockdiag(rel)) (bf16), bc = b @ blockdiag(rel) (f32)
// rel: [8][16][16] row-major. idx < 128*128 -> weight elem, row 128 -> bias elem.
__global__ void combine_rel(__bf16* __restrict__ Wct, float* __restrict__ bc,
                            const float* __restrict__ W, const float* __restrict__ b,
                            const float* __restrict__ rel) {
  int idx = blockIdx.x * blockDim.x + threadIdx.x;
  if (idx >= 129 * 128) return;
  int i  = idx >> 7;
  int je = idx & 127;
  int h = je >> 4, e2 = je & 15;
  const float* rh = rel + h * 256 + e2;    // rel[h][d][e2], stride 16 over d
  if (i < 128) {
    const float* wr = W + i * 128 + h * 16;
    float sacc = 0.f;
    #pragma unroll
    for (int d = 0; d < 16; ++d) sacc += wr[d] * rh[d * 16];
    Wct[je * 128 + i] = f2bf(sacc);        // transposed store: [col][k]
  } else {
    const float* br = b + h * 16;
    float sacc = 0.f;
    #pragma unroll
    for (int d = 0; d < 16; ++d) sacc += br[d] * rh[d * 16];
    bc[je] = sacc;
  }
}

// alpha[e,h] = scale * p_rel[h] * dot16(q[dst], k_rel[src]); fused segment-max
__global__ void edge_alpha(float* __restrict__ alpha, const float* __restrict__ qdt,
                           const float* __restrict__ krel, const int* __restrict__ src,
                           const int* __restrict__ dst, const float* __restrict__ p_rel,
                           float* __restrict__ amax, int E) {
  int idx = blockIdx.x * blockDim.x + threadIdx.x;
  if (idx >= E * HEADS) return;
  int e = idx >> 3, h = idx & 7;
  int s = src[e], d = dst[e];
  const float4* qp = reinterpret_cast<const float4*>(qdt + (size_t)d * DIM + h * HD);
  const float4* kp = reinterpret_cast<const float4*>(krel + (size_t)s * DIM + h * HD);
  float acc = 0.f;
  #pragma unroll
  for (int i = 0; i < 4; ++i) {
    float4 a = qp[i], b = kp[i];
    acc += a.x * b.x + a.y * b.y + a.z * b.z + a.w * b.w;
  }
  float v = acc * p_rel[h] * 0.25f;   // 1/sqrt(HD)
  alpha[idx] = v;
  atomicMaxF(amax + (size_t)d * HEADS + h, v);
}

// alpha <- exp(alpha - amax[dst]); denom[dst] += alpha
__global__ void edge_expsum(float* __restrict__ alpha, const float* __restrict__ amax,
                            float* __restrict__ denom, const int* __restrict__ dst, int E) {
  int idx = blockIdx.x * blockDim.x + threadIdx.x;
  if (idx >= E * HEADS) return;
  int e = idx >> 3, h = idx & 7;
  int d = dst[e];
  float v = __expf(alpha[idx] - amax[(size_t)d * HEADS + h]);
  alpha[idx] = v;
  atomicAdd(denom + (size_t)d * HEADS + h, v);
}

// acc[dst] += (alpha/denom) * v_rel[src]   (16 f32 atomics per (edge,head))
__global__ void edge_scatter(float* __restrict__ acc, const float* __restrict__ alpha,
                             const float* __restrict__ denom, const float* __restrict__ vrel,
                             const int* __restrict__ src, const int* __restrict__ dst, int E) {
  int idx = blockIdx.x * blockDim.x + threadIdx.x;
  if (idx >= E * HEADS) return;
  int e = idx >> 3, h = idx & 7;
  int s = src[e], d = dst[e];
  float w = alpha[idx] / (denom[(size_t)d * HEADS + h] + 1e-16f);
  const float4* vp = reinterpret_cast<const float4*>(vrel + (size_t)s * DIM + h * HD);
  float* ap = acc + (size_t)d * DIM + h * HD;
  #pragma unroll
  for (int i = 0; i < 4; ++i) {
    float4 v = vp[i];
    atomicAdd(ap + 4 * i + 0, v.x * w);
    atomicAdd(ap + 4 * i + 1, v.y * w);
    atomicAdd(ap + 4 * i + 2, v.z * w);
    atomicAdd(ap + 4 * i + 3, v.w * w);
  }
}

// gb = bf16(gelu_exact(acc))
__global__ void gelu_bf16(__bf16* __restrict__ gb, const float* __restrict__ acc, int n) {
  int i = blockIdx.x * blockDim.x + threadIdx.x;
  if (i < n) {
    float x = acc[i];
    float g = 0.5f * x * (1.f + erff(x * 0.70710678118654752f));
    gb[i] = f2bf(g);
  }
}

// ---------------------------------------------------------------------------
// host side
// ---------------------------------------------------------------------------
static const int NN[3]       = {50000, 30000, 100000};           // mirna, disease, mrna
static const size_t XOFF[3]  = {0, 50000ull * DIM, 80000ull * DIM};
static const int EST[9]      = {0, 1, 0, 2, 2, 1, 2, 0, 1};
static const int EDT[9]      = {0, 1, 2, 0, 1, 2, 2, 1, 0};
static const int EN[9]       = {300000, 200000, 400000, 400000, 400000,
                                400000, 400000, 400000, 400000};
static const int NTOT        = 180000;
static const int NMAX        = 100000;
static const int EMAX        = 400000;

struct LayerP {
  const float *k_w[3], *k_b[3], *q_w[3], *q_b[3], *v_w[3], *v_b[3],
              *a_w[3], *a_b[3], *skip[3], *a_rel[9], *m_rel[9], *p_rel[9];
};

// Map the 54 param leaves per layer. Two plausible flattenings are supported:
// (A) dict-insertion order; (B) jax tree_flatten (sorted keys).
// in_sizes[21]==16384 (a 128x128 weight) iff ordering (A).
static void map_params(void* const* d_in, const int* in_sizes, int layer, LayerP* P) {
  const int base = 21 + 54 * layer;
  auto F = [&](int i) { return (const float*)d_in[i]; };
  const bool orderA = (in_sizes[21] == DIM * DIM);
  if (orderA) {
    for (int t = 0; t < 3; ++t) {
      P->k_w[t] = F(base + 0 + t);  P->k_b[t] = F(base + 3 + t);
      P->q_w[t] = F(base + 6 + t);  P->q_b[t] = F(base + 9 + t);
      P->v_w[t] = F(base + 12 + t); P->v_b[t] = F(base + 15 + t);
      P->a_w[t] = F(base + 18 + t); P->a_b[t] = F(base + 21 + t);
      P->skip[t] = F(base + 24 + t);
    }
    for (int e = 0; e < 9; ++e) {
      P->a_rel[e] = F(base + 27 + e);
      P->m_rel[e] = F(base + 36 + e);
      P->p_rel[e] = F(base + 45 + e);
    }
  } else {
    static const int N2C[3] = {1, 0, 2};                    // sorted nt -> canonical
    static const int E2C[9] = {1, 8, 5, 7, 2, 0, 4, 6, 3};  // sorted ek -> canonical
    int i = base;
    for (int t = 0; t < 3; ++t) P->a_b[N2C[t]]  = F(i++);
    for (int e = 0; e < 9; ++e) P->a_rel[E2C[e]] = F(i++);
    for (int t = 0; t < 3; ++t) P->a_w[N2C[t]]  = F(i++);
    for (int t = 0; t < 3; ++t) P->k_b[N2C[t]]  = F(i++);
    for (int t = 0; t < 3; ++t) P->k_w[N2C[t]]  = F(i++);
    for (int e = 0; e < 9; ++e) P->m_rel[E2C[e]] = F(i++);
    for (int e = 0; e < 9; ++e) P->p_rel[E2C[e]] = F(i++);
    for (int t = 0; t < 3; ++t) P->q_b[N2C[t]]  = F(i++);
    for (int t = 0; t < 3; ++t) P->q_w[N2C[t]]  = F(i++);
    for (int t = 0; t < 3; ++t) P->skip[N2C[t]] = F(i++);
    for (int t = 0; t < 3; ++t) P->v_b[N2C[t]]  = F(i++);
    for (int t = 0; t < 3; ++t) P->v_w[N2C[t]]  = F(i++);
  }
}

extern "C" void kernel_launch(void* const* d_in, const int* in_sizes, int n_in,
                              void* d_out, int out_size, void* d_ws, size_t ws_size,
                              hipStream_t stream) {
  (void)n_in; (void)out_size; (void)ws_size;

  // ---- carve workspace (~450 MB) ----
  char* w = (char*)d_ws;
  auto carve = [&](size_t bytes) { void* p = (void*)w; w += (bytes + 255) & ~(size_t)255; return p; };
  float*  xbuf  = (float*)  carve((size_t)NTOT * DIM * 4);   // inter-layer activations
  float*  qbuf  = (float*)  carve((size_t)NTOT * DIM * 4);   // Q projections (f32, L2-resident)
  float*  accb  = (float*)  carve((size_t)NTOT * DIM * 4);   // aggregated messages
  __bf16* xb    = (__bf16*) carve((size_t)NTOT * DIM * 2);   // bf16 activations (also gelu out)
  float*  krel  = (float*)  carve((size_t)NMAX * DIM * 4);
  float*  vrel  = (float*)  carve((size_t)NMAX * DIM * 4);
  float*  alpha = (float*)  carve((size_t)EMAX * HEADS * 4);
  float*  amax  = (float*)  carve((size_t)NMAX * HEADS * 4);
  float*  denom = (float*)  carve((size_t)NMAX * HEADS * 4);
  __bf16* wq_bf = (__bf16*) carve((size_t)3 * DIM * DIM * 2); // q_w^T per node type
  __bf16* wa_bf = (__bf16*) carve((size_t)3 * DIM * DIM * 2); // a_w^T per node type
  __bf16* wk_bf = (__bf16*) carve((size_t)DIM * DIM * 2);     // fused key weight^T
  __bf16* wv_bf = (__bf16*) carve((size_t)DIM * DIM * 2);     // fused value weight^T
  float*  bk_c  = (float*)  carve(DIM * 4);
  float*  bv_c  = (float*)  carve(DIM * 4);

  const int* esrc[9]; const int* edst[9];
  for (int j = 0; j < 9; ++j) {
    esrc[j] = (const int*)d_in[3 + 2 * j];
    edst[j] = (const int*)d_in[4 + 2 * j];
  }

  for (int layer = 0; layer < 2; ++layer) {
    LayerP P; map_params(d_in, in_sizes, layer, &P);

    const float* xcur[3]; float* xnext[3];
    for (int t = 0; t < 3; ++t) {
      xcur[t]  = (layer == 0) ? (const float*)d_in[t] : (xbuf + XOFF[t]);
      xnext[t] = (layer == 0) ? (xbuf + XOFF[t]) : ((float*)d_out + XOFF[t]);
    }

    // 1) activations -> bf16; convert+transpose q/a weights
    for (int t = 0; t < 3; ++t) {
      int n = NN[t] * DIM;
      conv_bf16<<<CDIV(n, 256), 256, 0, stream>>>(xb + XOFF[t], xcur[t], n);
      conv_bf16_t<<<CDIV(DIM * DIM, 256), 256, 0, stream>>>(wq_bf + (size_t)t * DIM * DIM, P.q_w[t]);
      conv_bf16_t<<<CDIV(DIM * DIM, 256), 256, 0, stream>>>(wa_bf + (size_t)t * DIM * DIM, P.a_w[t]);
    }

    // 2) Q projections (WMMA)
    for (int t = 0; t < 3; ++t)
      hgt_gemm128<<<CDIV(NN[t], 64), 128, 0, stream>>>(wq_bf + (size_t)t * DIM * DIM, xb + XOFF[t],
                                                       P.q_b[t], qbuf + XOFF[t], NN[t], nullptr, nullptr);

    // 3) zero message accumulators
    fill_f32<<<CDIV(NTOT * DIM, 256), 256, 0, stream>>>(accb, 0.f, NTOT * DIM);

    // 4) per edge type: fused relation GEMMs + segment-softmax attention + scatter
    for (int j = 0; j < 9; ++j) {
      const int st = EST[j], dt = EDT[j], E = EN[j];
      combine_rel<<<CDIV(129 * 128, 256), 256, 0, stream>>>(wk_bf, bk_c, P.k_w[st], P.k_b[st], P.a_rel[j]);
      combine_rel<<<CDIV(129 * 128, 256), 256, 0, stream>>>(wv_bf, bv_c, P.v_w[st], P.v_b[st], P.m_rel[j]);
      hgt_gemm128<<<CDIV(NN[st], 64), 128, 0, stream>>>(wk_bf, xb + XOFF[st], bk_c, krel, NN[st], nullptr, nullptr);
      hgt_gemm128<<<CDIV(NN[st], 64), 128, 0, stream>>>(wv_bf, xb + XOFF[st], bv_c, vrel, NN[st], nullptr, nullptr);

      const int nd8 = NN[dt] * HEADS;
      fill_f32<<<CDIV(nd8, 256), 256, 0, stream>>>(amax, -INFINITY, nd8);
      fill_f32<<<CDIV(nd8, 256), 256, 0, stream>>>(denom, 0.f, nd8);

      const int eh = E * HEADS;
      edge_alpha<<<CDIV(eh, 256), 256, 0, stream>>>(alpha, qbuf + XOFF[dt], krel,
                                                    esrc[j], edst[j], P.p_rel[j], amax, E);
      edge_expsum<<<CDIV(eh, 256), 256, 0, stream>>>(alpha, amax, denom, edst[j], E);
      edge_scatter<<<CDIV(eh, 256), 256, 0, stream>>>(accb + XOFF[dt], alpha, denom, vrel,
                                                      esrc[j], edst[j], E);
    }

    // 5) gelu -> bf16 (reuse xb), output projection + sigmoid-skip blend (WMMA)
    for (int t = 0; t < 3; ++t) {
      int n = NN[t] * DIM;
      gelu_bf16<<<CDIV(n, 256), 256, 0, stream>>>(xb + XOFF[t], accb + XOFF[t], n);
      hgt_gemm128<<<CDIV(NN[t], 64), 128, 0, stream>>>(wa_bf + (size_t)t * DIM * DIM, xb + XOFF[t],
                                                       P.a_b[t], xnext[t], NN[t], xcur[t], P.skip[t]);
    }
  }
}
